// MaskedAutoencoderViT_39676907888590
// MI455X (gfx1250) — matrix operand order repaired and checked by reading, not compiled, not run on hardware
//
#include <hip/hip_runtime.h>
#include <math.h>

// ---------------------------------------------------------------------------
// MAE-ViT forward for MI455X (gfx1250): bf16 WMMA GEMMs + WMMA flash attention,
// with async global->LDS staging (ASYNCcnt) when the toolchain exposes it.
// ---------------------------------------------------------------------------

typedef __attribute__((ext_vector_type(8)))  float  v8f;
typedef __attribute__((ext_vector_type(16))) __bf16 v16bf;
typedef int v4i __attribute__((vector_size(16)));   // pointee type expected by
                                                    // global_load_async_to_lds_b128

union FragBF { uint4 q[2]; v16bf v; };

#define GTILE 128
#define GKB   32

#if __has_builtin(__builtin_amdgcn_global_load_async_to_lds_b128) && \
    __has_builtin(__builtin_amdgcn_s_wait_asynccnt)
#define HAVE_ASYNC_LDS 1
#else
#define HAVE_ASYNC_LDS 0
#endif

// ---------------------------------------------------------------------------
// Generic GEMM: C[M,N] = A[M,K] @ B[K,N] (+bias)(+residual)(+GELU)
// A,B,residual bf16; bias f32; C bf16 or f32 (flags&2 -> f32). flags&1 -> GELU.
// K % 32 == 0 required (true for every GEMM in this model).
// ---------------------------------------------------------------------------
__global__ __launch_bounds__(256) void gemm_bf16_wmma(
    const __bf16* __restrict__ A, int lda,
    const __bf16* __restrict__ B, int ldb,
    void* __restrict__ Cp, int ldc,
    const float* __restrict__ bias,
    const __bf16* __restrict__ residual, int ldres,
    int M, int Ncols, int K, int flags)
{
  __shared__ __align__(16) __bf16 As[2][GTILE * GKB];
  __shared__ __align__(16) __bf16 Bs[2][GTILE * GKB];

  const int tid  = threadIdx.x;
  const int lane = tid & 31;
  const int wave = tid >> 5;
  const int wm   = wave >> 2;      // 0..1  (64-row slab)
  const int wn   = wave & 3;       // 0..3  (32-col slab)
  const int m0   = blockIdx.y * GTILE;
  const int n0   = blockIdx.x * GTILE;
  const bool edgeM = (m0 + GTILE > M);

  v8f acc[4][2];
#pragma unroll
  for (int i = 0; i < 4; ++i)
#pragma unroll
    for (int j = 0; j < 2; ++j)
#pragma unroll
      for (int v = 0; v < 8; ++v) acc[i][j][v] = 0.0f;

  const int nK = K / GKB;

  // A tile: 128 rows x 32 cols; 2 threads per row, 32B each. Interior blocks
  // stream memory->LDS asynchronously (ASYNCcnt); edge blocks zero-pad in regs.
  auto stageA = [&](int buf, int kt) {
    const int k0 = kt * GKB;
    const int r = tid >> 1, h = tid & 1;
    const int gr = m0 + r;
#if HAVE_ASYNC_LDS
    if (!edgeM) {
      const __bf16* gp = A + (size_t)gr * lda + k0 + h * 16;
      __builtin_amdgcn_global_load_async_to_lds_b128(
          (v4i*)gp, (v4i*)&As[buf][r * GKB + h * 16], 0, 0);
      return;
    }
#endif
    uint4 d0 = {0, 0, 0, 0}, d1 = {0, 0, 0, 0};
    if (gr < M) {
      const uint4* gp = (const uint4*)(A + (size_t)gr * lda + k0 + h * 16);
      d0 = gp[0];
      d1 = gp[1];
    }
    uint4* sp = (uint4*)(&As[buf][r * GKB + h * 16]);
    sp[0] = d0;
    sp[1] = d1;
  };

  // B tile: read row-major 32Kx128N, store transposed Bs[n][k] (sync scatter:
  // transposed layout keeps every WMMA B-fragment a contiguous b128 LDS read).
  auto stageB = [&](int buf, int kt) {
    const int k0 = kt * GKB;
    const int k = tid >> 3, seg = tid & 7;
    union { uint4 q[2]; __bf16 e[16]; } u;
    u.q[0] = make_uint4(0, 0, 0, 0);
    u.q[1] = make_uint4(0, 0, 0, 0);
    const int gn0 = n0 + seg * 16;
    if (gn0 < Ncols) {
      const uint4* gp = (const uint4*)(B + (size_t)(k0 + k) * ldb + gn0);
      u.q[0] = gp[0];
      u.q[1] = gp[1];
    }
#pragma unroll
    for (int e = 0; e < 16; ++e)
      Bs[buf][(seg * 16 + e) * GKB + k] = u.e[e];
  };

  stageA(0, 0);
  stageB(0, 0);
#if HAVE_ASYNC_LDS
  __builtin_amdgcn_s_wait_asynccnt(0);
#endif
  __syncthreads();

  const int g   = lane >> 4;   // half-wave selector
  const int l16 = lane & 15;

  for (int kt = 0; kt < nK; ++kt) {
    const int buf = kt & 1;
    if (kt + 1 < nK) {
      stageA(buf ^ 1, kt + 1);
      stageB(buf ^ 1, kt + 1);
    }
    if (kt + 2 < nK) {   // gfx1250 global_prefetch for tiles two steps ahead
      __builtin_prefetch(A + (size_t)(m0 + (tid >> 1)) * lda + (kt + 2) * GKB, 0, 3);
      __builtin_prefetch(B + (size_t)((kt + 2) * GKB + (tid >> 3)) * ldb + n0, 0, 3);
    }

    FragBF a[4], b[2];
#pragma unroll
    for (int mf = 0; mf < 4; ++mf) {
      // A layout (contiguous k order, same as B): lane row = lane&15; halves 0..15 -> K=16g+h
      const __bf16* base = &As[buf][(wm * 64 + mf * 16 + l16) * GKB];
      a[mf].q[0] = *(const uint4*)(base + 16 * g);
      a[mf].q[1] = *(const uint4*)(base + 16 * g + 8);
    }
#pragma unroll
    for (int nf = 0; nf < 2; ++nf) {
      // B layout (contiguous k order, same as A): N = lane&15; lanes 0-15 hold K=0..15, lanes 16-31 K=16..31
      const __bf16* base = &Bs[buf][(wn * 32 + nf * 16 + l16) * GKB + 16 * g];
      b[nf].q[0] = ((const uint4*)base)[0];
      b[nf].q[1] = ((const uint4*)base)[1];
    }

#pragma unroll
    for (int mf = 0; mf < 4; ++mf)
#pragma unroll
      for (int nf = 0; nf < 2; ++nf)
        acc[mf][nf] = __builtin_amdgcn_wmma_f32_16x16x32_bf16(
            false, a[mf].v, false, b[nf].v, (short)0, acc[mf][nf], false, false);

#if HAVE_ASYNC_LDS
    __builtin_amdgcn_s_wait_asynccnt(0);   // async stage must land before barrier
#endif
    __syncthreads();
  }

  // Epilogue. C layout: row = v + 8*(lane>>4), col = lane&15 per 16x16 tile.
#pragma unroll
  for (int mf = 0; mf < 4; ++mf) {
#pragma unroll
    for (int nf = 0; nf < 2; ++nf) {
      const int col = n0 + wn * 32 + nf * 16 + l16;
      if (col >= Ncols) continue;
#pragma unroll
      for (int v = 0; v < 8; ++v) {
        const int row = m0 + wm * 64 + mf * 16 + g * 8 + v;
        if (row >= M) continue;
        float val = acc[mf][nf][v];
        if (bias)     val += bias[col];
        if (residual) val += (float)residual[(size_t)row * ldres + col];
        if (flags & 1) val = 0.5f * val * (1.0f + erff(val * 0.70710678118654752f));
        if (flags & 2) ((float*)Cp)[(size_t)row * ldc + col] = val;
        else           ((__bf16*)Cp)[(size_t)row * ldc + col] = (__bf16)val;
      }
    }
  }
}

// ---------------------------------------------------------------------------
// Wave-level WMMA flash attention (online softmax).
// qkv layout: [B, T, 3, nh, HD] bf16. Each wave owns a 16-query tile; a block
// (8 waves) covers 128 queries. K chunk staged row-major in LDS (async for
// full chunks), V staged transposed so WMMA B-fragments are b128 LDS reads.
// ---------------------------------------------------------------------------
template <int HD>
__global__ __launch_bounds__(256) void attn_wmma_kernel(
    const __bf16* __restrict__ qkv, __bf16* __restrict__ out,
    int T, int nh, float scale)
{
  constexpr int NC = HD / 32;    // contraction fragments for S = Q*K^T
  constexpr int NT = HD / 16;    // output column tiles
  constexpr int CK = 64;         // keys per staged chunk

  __shared__ __align__(16) __bf16 Ks[CK * HD];       // [key][hd]
  __shared__ __align__(16) __bf16 Vt[HD * CK];       // [hd][key]  (transposed)
  __shared__ __align__(16) __bf16 Pb[8][16 * 32];    // per-wave P scratch

  const int b   = blockIdx.z;
  const int h   = blockIdx.y;
  const int dm  = nh * HD;
  const int tid = threadIdx.x;
  const int lane = tid & 31, wave = tid >> 5;
  const int g = lane >> 4, l16 = lane & 15;
  const int q0 = blockIdx.x * 128 + wave * 16;

  // --- Q fragments (A layout: row = lane&15, halves -> K = 16g+h) ---
  FragBF aq[NC];
#pragma unroll
  for (int c = 0; c < NC; ++c) {
    aq[c].q[0] = make_uint4(0, 0, 0, 0);
    aq[c].q[1] = make_uint4(0, 0, 0, 0);
  }
  {
    const int qrow = q0 + l16;
    if (qrow < T) {
      const __bf16* qp = qkv + ((size_t)(b * T + qrow) * 3 + 0) * dm + h * HD;
#pragma unroll
      for (int c = 0; c < NC; ++c) {
        aq[c].q[0] = *(const uint4*)(qp + c * 32 + 16 * g);
        aq[c].q[1] = *(const uint4*)(qp + c * 32 + 16 * g + 8);
      }
    }
  }

  v8f o[NT];
#pragma unroll
  for (int nt = 0; nt < NT; ++nt)
#pragma unroll
    for (int v = 0; v < 8; ++v) o[nt][v] = 0.0f;
  float m[8], l[8];
#pragma unroll
  for (int v = 0; v < 8; ++v) { m[v] = -1e30f; l[v] = 0.0f; }

  for (int kc = 0; kc < T; kc += CK) {
    const bool fullChunk = (kc + CK <= T);
    // --- stage K rows (async when full) + V transposed (sync scatter) ---
    for (int i = tid * 8; i < CK * HD; i += 256 * 8) {
      const int kr = i / HD, col = i % HD;
      const int key = kc + kr;
#if HAVE_ASYNC_LDS
      if (fullChunk) {
        const __bf16* gk = qkv + ((size_t)(b * T + key) * 3 + 1) * dm + h * HD + col;
        __builtin_amdgcn_global_load_async_to_lds_b128(
            (v4i*)gk, (v4i*)&Ks[i], 0, 0);
      } else
#endif
      {
        uint4 dk = {0, 0, 0, 0};
        if (key < T)
          dk = *(const uint4*)(qkv + ((size_t)(b * T + key) * 3 + 1) * dm + h * HD + col);
        *(uint4*)(&Ks[i]) = dk;
      }
      uint4 dv = {0, 0, 0, 0};
      if (key < T)
        dv = *(const uint4*)(qkv + ((size_t)(b * T + key) * 3 + 2) * dm + h * HD + col);
      union { uint4 q; __bf16 e[8]; } u;
      u.q = dv;
#pragma unroll
      for (int e = 0; e < 8; ++e) Vt[(col + e) * CK + kr] = u.e[e];
    }
#if HAVE_ASYNC_LDS
    __builtin_amdgcn_s_wait_asynccnt(0);
#endif
    __syncthreads();

#pragma unroll
    for (int sub = 0; sub < 2; ++sub) {              // 32 keys per step
      // --- S = Q @ K^T for two 16-key tiles (WMMA) ---
      v8f s[2];
#pragma unroll
      for (int jt = 0; jt < 2; ++jt) {
#pragma unroll
        for (int v = 0; v < 8; ++v) s[jt][v] = 0.0f;
#pragma unroll
        for (int c = 0; c < NC; ++c) {
          FragBF bk;  // B[hd][key]: lane = key, halves = hd (contiguous in Ks row)
          const __bf16* base = &Ks[((sub * 2 + jt) * 16 + l16) * HD + c * 32 + 16 * g];
          bk.q[0] = ((const uint4*)base)[0];
          bk.q[1] = ((const uint4*)base)[1];
          s[jt] = __builtin_amdgcn_wmma_f32_16x16x32_bf16(
              false, aq[c].v, false, bk.v, (short)0, s[jt], false, false);
        }
      }

      // --- online softmax on WMMA C-layout (row = v+8g, col = l16) ---
#pragma unroll
      for (int v = 0; v < 8; ++v) {
        const float s0 = s[0][v] * scale;
        const float s1 = s[1][v] * scale;
        const int key0 = kc + sub * 32 + l16;
        const int key1 = key0 + 16;
        const bool k0ok = key0 < T, k1ok = key1 < T;
        float loc = fmaxf(k0ok ? s0 : -1e30f, k1ok ? s1 : -1e30f);
#pragma unroll
        for (int d = 1; d < 16; d <<= 1) loc = fmaxf(loc, __shfl_xor(loc, d, 32));
        const float mn = fmaxf(m[v], loc);
        const float corr = __expf(m[v] - mn);
        const float p0 = k0ok ? __expf(s0 - mn) : 0.0f;
        const float p1 = k1ok ? __expf(s1 - mn) : 0.0f;
        float ps = p0 + p1;
#pragma unroll
        for (int d = 1; d < 16; d <<= 1) ps += __shfl_xor(ps, d, 32);
        l[v] = l[v] * corr + ps;
        m[v] = mn;
#pragma unroll
        for (int nt = 0; nt < NT; ++nt) o[nt][v] *= corr;
        // P: C-layout -> wave-private LDS tile [qrow][key32]
        Pb[wave][(v + 8 * g) * 32 + l16]      = (__bf16)p0;
        Pb[wave][(v + 8 * g) * 32 + 16 + l16] = (__bf16)p1;
      }

      // --- reload P as A fragment (same wave; LDS ops are in-order) ---
      FragBF pa;
      const __bf16* pbase = &Pb[wave][l16 * 32];
      pa.q[0] = *(const uint4*)(pbase + 16 * g);
      pa.q[1] = *(const uint4*)(pbase + 16 * g + 8);

      // --- O += P @ V  (WMMA; B from transposed V tile) ---
#pragma unroll
      for (int nt = 0; nt < NT; ++nt) {
        FragBF bv;
        const __bf16* vb = &Vt[(nt * 16 + l16) * CK + sub * 32 + 16 * g];
        bv.q[0] = ((const uint4*)vb)[0];
        bv.q[1] = ((const uint4*)vb)[1];
        o[nt] = __builtin_amdgcn_wmma_f32_16x16x32_bf16(
            false, pa.v, false, bv.v, (short)0, o[nt], false, false);
      }
    }
    __syncthreads();
  }

  // --- normalize and store (C layout: row = v+8g, col = nt*16+l16) ---
#pragma unroll
  for (int v = 0; v < 8; ++v) {
    const int qrow = q0 + v + 8 * g;
    if (qrow >= T) continue;
    const float inv = 1.0f / l[v];
    __bf16* op = out + (size_t)(b * T + qrow) * dm + h * HD;
#pragma unroll
    for (int nt = 0; nt < NT; ++nt) op[nt * 16 + l16] = (__bf16)(o[nt][v] * inv);
  }
}

// ---------------------------------------------------------------------------
// LayerNorm over rows of length D; bf16 in/out, f32 gamma/beta.
// ---------------------------------------------------------------------------
__global__ __launch_bounds__(256) void layernorm_kernel(
    const __bf16* __restrict__ x, __bf16* __restrict__ y,
    const float* __restrict__ w, const float* __restrict__ b, int D)
{
  __shared__ float red[256];
  const int row = blockIdx.x;
  const __bf16* xr = x + (size_t)row * D;

  float s = 0.0f;
  for (int d = threadIdx.x; d < D; d += 256) s += (float)xr[d];
  red[threadIdx.x] = s;
  __syncthreads();
  for (int o = 128; o > 0; o >>= 1) {
    if (threadIdx.x < o) red[threadIdx.x] += red[threadIdx.x + o];
    __syncthreads();
  }
  const float mean = red[0] / (float)D;
  __syncthreads();

  float v = 0.0f;
  for (int d = threadIdx.x; d < D; d += 256) {
    const float tdiff = (float)xr[d] - mean;
    v += tdiff * tdiff;
  }
  red[threadIdx.x] = v;
  __syncthreads();
  for (int o = 128; o > 0; o >>= 1) {
    if (threadIdx.x < o) red[threadIdx.x] += red[threadIdx.x + o];
    __syncthreads();
  }
  const float rstd = rsqrtf(red[0] / (float)D + 1e-5f);

  __bf16* yr = y + (size_t)row * D;
  for (int d = threadIdx.x; d < D; d += 256)
    yr[d] = (__bf16)(((float)xr[d] - mean) * rstd * w[d] + b[d]);
}

// ---------------------------------------------------------------------------
// Small helpers
// ---------------------------------------------------------------------------
__global__ void cvt_f32_bf16_kernel(const float* __restrict__ in,
                                    __bf16* __restrict__ out, long long n)
{
  const long long i = (long long)blockIdx.x * 256 + threadIdx.x;
  if (i < n) out[i] = (__bf16)in[i];
}

// imgs [8,6,224,224] -> pt [8,196,6,256] (f32 target + bf16 GEMM operand)
__global__ void patchify_kernel(const float* __restrict__ imgs,
                                float* __restrict__ target,
                                __bf16* __restrict__ ptb)
{
  const long long idx = (long long)blockIdx.x * 256 + threadIdx.x;
  if (idx >= (long long)8 * 196 * 6 * 256) return;
  const int qq = (int)(idx & 255);
  long long r = idx >> 8;
  const int c = (int)(r % 6); r /= 6;
  const int p = (int)(r % 196);
  const int n = (int)(r / 196);
  const int ph = p / 14, pw = p % 14, py = qq / 16, px = qq % 16;
  const float v = imgs[(((size_t)n * 6 + c) * 224 + (ph * 16 + py)) * 224 + (pw * 16 + px)];
  target[idx] = v;
  ptb[idx] = (__bf16)v;
}

// x[n,l,d] += pos_embed[l+1,d]   (x bf16 [8,1176,768], pos f32)
__global__ void add_pos_kernel(__bf16* __restrict__ x, const float* __restrict__ pos)
{
  const long long idx = (long long)blockIdx.x * 256 + threadIdx.x;
  if (idx >= (long long)8 * 1176 * 768) return;
  const int d = (int)(idx % 768);
  const int l = (int)((idx / 768) % 1176);
  x[idx] = (__bf16)((float)x[idx] + pos[(size_t)(l + 1) * 768 + d]);
}

// Per-patch stable argsort of 6 noise values; keep first 3. Writes kept ids
// and the f32 mask (1 = masked) straight into d_out's mask region.
__global__ void mask_kernel(const float* __restrict__ noise,
                            int* __restrict__ kept, float* __restrict__ mask_out)
{
  const int idx = blockIdx.x * 256 + threadIdx.x;   // n*196+p
  if (idx >= 8 * 196) return;
  const int n = idx / 196, p = idx % 196;
  const float* nz = noise + (size_t)idx * 6;
  int ord[6];
  float vals[6];
  for (int c = 0; c < 6; ++c) { ord[c] = c; vals[c] = nz[c]; }
  for (int i = 1; i < 6; ++i) {           // stable insertion sort
    const int oi = ord[i];
    const float vi = vals[oi];
    int j = i - 1;
    while (j >= 0 && vals[ord[j]] > vi) { ord[j + 1] = ord[j]; --j; }
    ord[j + 1] = oi;
  }
  for (int i = 0; i < 3; ++i) kept[n * 588 + p * 3 + i] = p * 6 + ord[i];
  for (int i = 0; i < 6; ++i)
    mask_out[(size_t)n * 1176 + p * 6 + ord[i]] = (i < 3) ? 0.0f : 1.0f;
}

// ids_restore: inverse of [kept | masked-ascending]
__global__ void restore_kernel(const int* __restrict__ kept,
                               const float* __restrict__ mask,
                               int* __restrict__ ids_restore)
{
  const int n = blockIdx.x;
  for (int i = 0; i < 588; ++i) ids_restore[n * 1176 + kept[n * 588 + i]] = i;
  int cnt = 0;
  for (int l = 0; l < 1176; ++l)
    if (mask[n * 1176 + l] > 0.5f) ids_restore[n * 1176 + l] = 588 + cnt++;
}

// xe[n,0]=cls+pos0 ; xe[n,1+j]=x[n,kept[j]]  (x already has pos added)
__global__ void gather_enc_kernel(const __bf16* __restrict__ x,
                                  const int* __restrict__ kept,
                                  const float* __restrict__ cls,
                                  const float* __restrict__ pos,
                                  __bf16* __restrict__ xe)
{
  const long long idx = (long long)blockIdx.x * 256 + threadIdx.x;
  if (idx >= (long long)8 * 589 * 768) return;
  const int d = (int)(idx % 768);
  const long long r = idx / 768;
  const int row = (int)(r % 589);
  const int n = (int)(r / 589);
  float v;
  if (row == 0) v = cls[d] + pos[d];
  else v = (float)x[((size_t)n * 1176 + kept[n * 588 + row - 1]) * 768 + d];
  xe[idx] = (__bf16)v;
}

// Re-insert mask tokens per ids_restore, add decoder pos embed.
__global__ void build_dec_kernel(const __bf16* __restrict__ xd,
                                 const int* __restrict__ ids_restore,
                                 const float* __restrict__ mask_token,
                                 const float* __restrict__ dpos,
                                 __bf16* __restrict__ xf)
{
  const long long idx = (long long)blockIdx.x * 256 + threadIdx.x;
  if (idx >= (long long)8 * 1177 * 512) return;
  const int d = (int)(idx % 512);
  const long long r = idx / 512;
  const int l = (int)(r % 1177);
  const int n = (int)(r / 1177);
  float v;
  if (l == 0) v = (float)xd[((size_t)n * 589) * 512 + d];
  else {
    const int rr = ids_restore[n * 1176 + l - 1];
    v = (rr < 588) ? (float)xd[((size_t)n * 589 + 1 + rr) * 512 + d] : mask_token[d];
  }
  xf[idx] = (__bf16)(v + dpos[(size_t)l * 512 + d]);
}

// Per-token squared error (mean over 256), write pred f32, masked partials.
__global__ __launch_bounds__(256) void loss_token_kernel(
    const float* __restrict__ predbuf, const float* __restrict__ target,
    const float* __restrict__ mask, float* __restrict__ pred_out,
    float* __restrict__ partial)
{
  __shared__ float red[256];
  const int tok = blockIdx.x;               // 0..9407  (n*1176+l)
  const int n = tok / 1176, l = tok % 1176;
  const int d = threadIdx.x;
  const float pv = predbuf[((size_t)n * 1177 + 1 + l) * 256 + d];
  const float tg = target[(size_t)tok * 256 + d];
  pred_out[(size_t)tok * 256 + d] = pv;
  const float diff = pv - tg;
  red[d] = diff * diff;
  __syncthreads();
  for (int o = 128; o > 0; o >>= 1) {
    if (d < o) red[d] += red[d + o];
    __syncthreads();
  }
  if (d == 0) partial[tok] = (red[0] / 256.0f) * mask[tok];
}

__global__ void loss_reduce_kernel(const float* __restrict__ partial,
                                   float* __restrict__ loss_out)
{
  __shared__ float red[256];
  float s = 0.0f;
  for (int i = threadIdx.x; i < 9408; i += 256) s += partial[i];
  red[threadIdx.x] = s;
  __syncthreads();
  for (int o = 128; o > 0; o >>= 1) {
    if (threadIdx.x < o) red[threadIdx.x] += red[threadIdx.x + o];
    __syncthreads();
  }
  // mask.sum() == N*NP*(C-K) == 4704 exactly (fixed keep count per patch)
  if (threadIdx.x == 0) loss_out[0] = red[0] / 4704.0f;
}

// ---------------------------------------------------------------------------
// Host orchestration
// ---------------------------------------------------------------------------
extern "C" void kernel_launch(void* const* d_in, const int* in_sizes, int n_in,
                              void* d_out, int out_size, void* d_ws, size_t ws_size,
                              hipStream_t stream)
{
  (void)in_sizes; (void)n_in; (void)out_size; (void)ws_size;

  // --- input map (setup_inputs insertion order, dicts flattened in order) ---
  const float* imgs    = (const float*)d_in[0];
  const float* noise   = (const float*)d_in[1];
  const float* patch_w = (const float*)d_in[2];
  const float* patch_b = (const float*)d_in[3];
  const float* cls     = (const float*)d_in[4];
  const float* pos     = (const float*)d_in[5];
  const float* e_ln1w  = (const float*)d_in[6];
  const float* e_ln1b  = (const float*)d_in[7];
  const float* e_qkvw  = (const float*)d_in[8];
  const float* e_qkvb  = (const float*)d_in[9];
  const float* e_projw = (const float*)d_in[10];
  const float* e_projb = (const float*)d_in[11];
  const float* e_ln2w  = (const float*)d_in[12];
  const float* e_ln2b  = (const float*)d_in[13];
  const float* e_fc1w  = (const float*)d_in[14];
  const float* e_fc1b  = (const float*)d_in[15];
  const float* e_fc2w  = (const float*)d_in[16];
  const float* e_fc2b  = (const float*)d_in[17];
  const float* normw   = (const float*)d_in[18];
  const float* normb   = (const float*)d_in[19];
  const float* dew     = (const float*)d_in[20];
  const float* deb     = (const float*)d_in[21];
  const float* mtok    = (const float*)d_in[22];
  const float* dpos    = (const float*)d_in[23];
  const float* d_ln1w  = (const float*)d_in[24];
  const float* d_ln1b  = (const float*)d_in[25];
  const float* d_qkvw  = (const float*)d_in[26];
  const float* d_qkvb  = (const float*)d_in[27];
  const float* d_projw = (const float*)d_in[28];
  const float* d_projb = (const float*)d_in[29];
  const float* d_ln2w  = (const float*)d_in[30];
  const float* d_ln2b  = (const float*)d_in[31];
  const float* d_fc1w  = (const float*)d_in[32];
  const float* d_fc1b  = (const float*)d_in[33];
  const float* d_fc2w  = (const float*)d_in[34];
  const float* d_fc2b  = (const float*)d_in[35];
  const float* dnormw  = (const float*)d_in[36];
  const float* dnormb  = (const float*)d_in[37];
  const float* predw   = (const float*)d_in[38];
  const float* predb   = (const float*)d_in[39];

  // --- output regions: [loss(1) | pred(8*1176*256) | mask(8*1176)] f32 ---
  float* out_loss = (float*)d_out;
  float* out_pred = out_loss + 1;
  float* out_mask = out_pred + (size_t)8 * 1176 * 256;

  // --- workspace bump allocator ---
  char* wp = (char*)d_ws;
  auto alloc = [&](size_t bytes) -> char* {
    char* p = wp;
    wp += (bytes + 255) & ~(size_t)255;
    return p;
  };

  const int ME = 8 * 589;    // 4712 encoder rows
  const int MD = 8 * 1177;   // 9416 decoder rows

  float*  tgt     = (float*)  alloc((size_t)8 * 1176 * 256 * 4);
  __bf16* ptb     = (__bf16*) alloc((size_t)8 * 1176 * 256 * 2);
  __bf16* xb      = (__bf16*) alloc((size_t)8 * 1176 * 768 * 2);
  int*    kept    = (int*)    alloc((size_t)8 * 588 * 4);
  int*    idsr    = (int*)    alloc((size_t)8 * 1176 * 4);
  __bf16* xeA     = (__bf16*) alloc((size_t)ME * 768 * 2);
  __bf16* xeB     = (__bf16*) alloc((size_t)ME * 768 * 2);
  __bf16* ybuf    = (__bf16*) alloc((size_t)ME * 768 * 2);
  __bf16* qkvE    = (__bf16*) alloc((size_t)ME * 2304 * 2);
  __bf16* abuf    = (__bf16*) alloc((size_t)ME * 768 * 2);
  __bf16* hbuf    = (__bf16*) alloc((size_t)ME * 3072 * 2);
  __bf16* xdsm    = (__bf16*) alloc((size_t)ME * 512 * 2);
  __bf16* xdA     = (__bf16*) alloc((size_t)MD * 512 * 2);
  __bf16* xdB     = (__bf16*) alloc((size_t)MD * 512 * 2);
  __bf16* ydec    = (__bf16*) alloc((size_t)MD * 512 * 2);
  __bf16* qkvD    = (__bf16*) alloc((size_t)MD * 1536 * 2);
  __bf16* adec    = (__bf16*) alloc((size_t)MD * 512 * 2);
  __bf16* hdec    = (__bf16*) alloc((size_t)MD * 2048 * 2);
  float*  predbuf = (float*)  alloc((size_t)MD * 256 * 4);
  float*  partial = (float*)  alloc((size_t)9408 * 4);
  // bf16 weight copies
  __bf16* pw_b    = (__bf16*) alloc((size_t)6 * 256 * 768 * 2);
  __bf16* eqkv_b  = (__bf16*) alloc((size_t)12 * 768 * 2304 * 2);
  __bf16* eproj_b = (__bf16*) alloc((size_t)12 * 768 * 768 * 2);
  __bf16* efc1_b  = (__bf16*) alloc((size_t)12 * 768 * 3072 * 2);
  __bf16* efc2_b  = (__bf16*) alloc((size_t)12 * 3072 * 768 * 2);
  __bf16* dew_b   = (__bf16*) alloc((size_t)768 * 512 * 2);
  __bf16* dqkv_b  = (__bf16*) alloc((size_t)4 * 512 * 1536 * 2);
  __bf16* dproj_b = (__bf16*) alloc((size_t)4 * 512 * 512 * 2);
  __bf16* dfc1_b  = (__bf16*) alloc((size_t)4 * 512 * 2048 * 2);
  __bf16* dfc2_b  = (__bf16*) alloc((size_t)4 * 2048 * 512 * 2);
  __bf16* predw_b = (__bf16*) alloc((size_t)512 * 256 * 2);

  auto cvt = [&](const float* src, __bf16* dst, long long n) {
    cvt_f32_bf16_kernel<<<dim3((unsigned)((n + 255) / 256)), 256, 0, stream>>>(src, dst, n);
  };
  auto gemm = [&](const __bf16* A, int lda, const __bf16* B, int ldb,
                  void* C, int ldc, const float* bias,
                  const __bf16* res, int ldres, int M, int Nc, int K, int flags) {
    dim3 grid((Nc + GTILE - 1) / GTILE, (M + GTILE - 1) / GTILE);
    gemm_bf16_wmma<<<grid, 256, 0, stream>>>(A, lda, B, ldb, C, ldc,
                                             bias, res, ldres, M, Nc, K, flags);
  };

  // --- weight conversion (f32 -> bf16) ---
  cvt(patch_w, pw_b,    (long long)6 * 256 * 768);
  cvt(e_qkvw,  eqkv_b,  (long long)12 * 768 * 2304);
  cvt(e_projw, eproj_b, (long long)12 * 768 * 768);
  cvt(e_fc1w,  efc1_b,  (long long)12 * 768 * 3072);
  cvt(e_fc2w,  efc2_b,  (long long)12 * 3072 * 768);
  cvt(dew,     dew_b,   (long long)768 * 512);
  cvt(d_qkvw,  dqkv_b,  (long long)4 * 512 * 1536);
  cvt(d_projw, dproj_b, (long long)4 * 512 * 512);
  cvt(d_fc1w,  dfc1_b,  (long long)4 * 512 * 2048);
  cvt(d_fc2w,  dfc2_b,  (long long)4 * 2048 * 512);
  cvt(predw,   predw_b, (long long)512 * 256);

  // --- patchify + per-channel patch embed (6 strided WMMA GEMMs) ---
  {
    const long long tot = (long long)8 * 1176 * 256;
    patchify_kernel<<<dim3((unsigned)((tot + 255) / 256)), 256, 0, stream>>>(imgs, tgt, ptb);
  }
  for (int c = 0; c < 6; ++c) {
    gemm(ptb + (size_t)c * 256, 6 * 256,
         pw_b + (size_t)c * 256 * 768, 768,
         xb + (size_t)c * 768, 6 * 768,
         patch_b + (size_t)c * 768, nullptr, 0,
         8 * 196, 768, 256, /*flags=*/0);
  }
  {
    const long long tot = (long long)8 * 1176 * 768;
    add_pos_kernel<<<dim3((unsigned)((tot + 255) / 256)), 256, 0, stream>>>(xb, pos);
  }

  // --- masking / token selection ---
  mask_kernel<<<dim3((8 * 196 + 255) / 256), 256, 0, stream>>>(noise, kept, out_mask);
  restore_kernel<<<dim3(8), 1, 0, stream>>>(kept, out_mask, idsr);
  {
    const long long tot = (long long)8 * 589 * 768;
    gather_enc_kernel<<<dim3((unsigned)((tot + 255) / 256)), 256, 0, stream>>>(
        xb, kept, cls, pos, xeA);
  }

  // --- encoder: 12 transformer blocks (D=768, H=12, hd=64, T=589) ---
  __bf16* cur = xeA;
  __bf16* alt = xeB;
  for (int l = 0; l < 12; ++l) {
    layernorm_kernel<<<dim3(ME), 256, 0, stream>>>(cur, ybuf,
        e_ln1w + (size_t)l * 768, e_ln1b + (size_t)l * 768, 768);
    gemm(ybuf, 768, eqkv_b + (size_t)l * 768 * 2304, 2304, qkvE, 2304,
         e_qkvb + (size_t)l * 2304, nullptr, 0, ME, 2304, 768, 0);
    attn_wmma_kernel<64><<<dim3((589 + 127) / 128, 12, 8), 256, 0, stream>>>(
        qkvE, abuf, 589, 12, 0.125f);
    gemm(abuf, 768, eproj_b + (size_t)l * 768 * 768, 768, alt, 768,
         e_projb + (size_t)l * 768, cur, 768, ME, 768, 768, 0);
    { __bf16* t = cur; cur = alt; alt = t; }
    layernorm_kernel<<<dim3(ME), 256, 0, stream>>>(cur, ybuf,
        e_ln2w + (size_t)l * 768, e_ln2b + (size_t)l * 768, 768);
    gemm(ybuf, 768, efc1_b + (size_t)l * 768 * 3072, 3072, hbuf, 3072,
         e_fc1b + (size_t)l * 3072, nullptr, 0, ME, 3072, 768, /*GELU*/1);
    gemm(hbuf, 3072, efc2_b + (size_t)l * 3072 * 768, 768, alt, 768,
         e_fc2b + (size_t)l * 768, cur, 768, ME, 768, 3072, 0);
    { __bf16* t = cur; cur = alt; alt = t; }
  }

  // --- final encoder norm -> latent, decoder embed ---
  layernorm_kernel<<<dim3(ME), 256, 0, stream>>>(cur, ybuf, normw, normb, 768);
  gemm(ybuf, 768, dew_b, 512, xdsm, 512, deb, nullptr, 0, ME, 512, 768, 0);

  // --- re-insert mask tokens + decoder pos embed ---
  {
    const long long tot = (long long)8 * 1177 * 512;
    build_dec_kernel<<<dim3((unsigned)((tot + 255) / 256)), 256, 0, stream>>>(
        xdsm, idsr, mtok, dpos, xdA);
  }

  // --- decoder: 4 transformer blocks (DD=512, 16 heads, hd=32, T=1177) ---
  __bf16* dcur = xdA;
  __bf16* dalt = xdB;
  for (int l = 0; l < 4; ++l) {
    layernorm_kernel<<<dim3(MD), 256, 0, stream>>>(dcur, ydec,
        d_ln1w + (size_t)l * 512, d_ln1b + (size_t)l * 512, 512);
    gemm(ydec, 512, dqkv_b + (size_t)l * 512 * 1536, 1536, qkvD, 1536,
         d_qkvb + (size_t)l * 1536, nullptr, 0, MD, 1536, 512, 0);
    attn_wmma_kernel<32><<<dim3((1177 + 127) / 128, 16, 8), 256, 0, stream>>>(
        qkvD, adec, 1177, 16, 0.17677669529663687f);
    gemm(adec, 512, dproj_b + (size_t)l * 512 * 512, 512, dalt, 512,
         d_projb + (size_t)l * 512, dcur, 512, MD, 512, 512, 0);
    { __bf16* t = dcur; dcur = dalt; dalt = t; }
    layernorm_kernel<<<dim3(MD), 256, 0, stream>>>(dcur, ydec,
        d_ln2w + (size_t)l * 512, d_ln2b + (size_t)l * 512, 512);
    gemm(ydec, 512, dfc1_b + (size_t)l * 512 * 2048, 2048, hdec, 2048,
         d_fc1b + (size_t)l * 2048, nullptr, 0, MD, 2048, 512, /*GELU*/1);
    gemm(hdec, 2048, dfc2_b + (size_t)l * 2048 * 512, 512, dalt, 512,
         d_fc2b + (size_t)l * 512, dcur, 512, MD, 512, 2048, 0);
    { __bf16* t = dcur; dcur = dalt; dalt = t; }
  }

  // --- decoder norm + prediction head (f32 output) ---
  layernorm_kernel<<<dim3(MD), 256, 0, stream>>>(dcur, ydec, dnormw, dnormb, 512);
  gemm(ydec, 512, predw_b, 256, predbuf, 256, predb, nullptr, 0,
       MD, 256, 512, /*store f32*/2);

  // --- loss + write pred to d_out ---
  loss_token_kernel<<<dim3(9408), 256, 0, stream>>>(predbuf, tgt, out_mask,
                                                    out_pred, partial);
  loss_reduce_kernel<<<dim3(1), 256, 0, stream>>>(partial, out_loss);
}
